// WTConv1d_6906307412084
// MI455X (gfx1250) — matrix operand order repaired and checked by reading, not compiled, not run on hardware
//
#include <hip/hip_runtime.h>
#include <hip/hip_bf16.h>

// ---------------------------------------------------------------------------
// WTConv1d for MI455X (gfx1250, wave32, WMMA).
//   out = conv1d_same(x, conv_w) + bias + scale * wavelet_rec(x)
// Main conv = GEMM  M=256, K=1280, N=131072  -> v_wmma_f32_16x16x32_f16,
// f32 accumulate.  Wavelet branch = fp32, LDS-tiled, bandwidth-bound.
// Interior l-blocks use an unconditional-load GEMM path (no EXEC juggling,
// no fragment zeroing); only the 2 edge l-blocks run the guarded path.
// ---------------------------------------------------------------------------

#define CC 256
#define LL 16384
#define BB 8

typedef __attribute__((ext_vector_type(16))) _Float16 v16h;
typedef __attribute__((ext_vector_type(8)))  _Float16 v8h;
typedef __attribute__((ext_vector_type(8)))  float    v8f;

union V16 { v16h v; v8h h[2]; };

#define SQf 0.7071067811865476f

// ---------------------------------------------------------------------------
// Pack conv_w (C,C,5) fp32 -> wf16[k][co][ci]  (row-major in ci)
// ---------------------------------------------------------------------------
__global__ void __launch_bounds__(256)
pack_w_kernel(const float* __restrict__ w, _Float16* __restrict__ wp)
{
    int i = blockIdx.x * 256 + threadIdx.x;          // 5*256*256 = 327680
    if (i < 5 * CC * CC) {
        int k  = i >> 16;
        int co = (i >> 8) & 255;
        int ci = i & 255;
        wp[i] = (_Float16)w[((size_t)co * CC + ci) * 5 + k];
    }
}

// ---------------------------------------------------------------------------
// Pack x (B,C,L) fp32 -> xT (B,L,C) f16 via LDS 32x32 tile transpose.
// ---------------------------------------------------------------------------
__global__ void __launch_bounds__(256)
pack_x_kernel(const float* __restrict__ x, _Float16* __restrict__ xT)
{
    __shared__ float tile[32][33];
    const int l0 = blockIdx.x * 32;
    const int c0 = blockIdx.y * 32;
    const int b  = blockIdx.z;
    const int tx = threadIdx.x & 31;
    const int ty = threadIdx.x >> 5;                  // 0..7

    const float* xb = x + ((size_t)b * CC + c0) * LL + l0;
#pragma unroll
    for (int i = 0; i < 4; ++i)
        tile[ty + i * 8][tx] = xb[(size_t)(ty + i * 8) * LL + tx];
    __syncthreads();

    _Float16* xo = xT + ((size_t)b * LL + l0) * CC + c0;
#pragma unroll
    for (int i = 0; i < 4; ++i)
        xo[(size_t)(ty + i * 8) * CC + tx] = (_Float16)tile[tx][ty + i * 8];
}

// ---------------------------------------------------------------------------
// Wavelet branch: 3-level Haar DWT + depthwise conv5 on details + inverse.
// One workgroup per (b, c, tile of 2048 samples), 64-sample halo, all in LDS.
// ---------------------------------------------------------------------------
#define WT_T 2048
#define WT_H 64
#define WT_E (WT_T + 2 * WT_H)                        // 2176

__device__ __forceinline__ float conv5(const float* d, int m, int len,
                                       const float wc[5])
{
    float s = 0.f;
#pragma unroll
    for (int j = 0; j < 5; ++j) {
        int p = m - 2 + j;
        float v = (p >= 0 && p < len) ? d[p] : 0.f;
        s += v * wc[j];
    }
    return s;
}

__global__ void __launch_bounds__(256)
wavelet_rec_kernel(const float* __restrict__ x,
                   const float* __restrict__ w0, const float* __restrict__ w1,
                   const float* __restrict__ w2, float* __restrict__ rec)
{
    __shared__ float sx[WT_E];
    __shared__ float sa1[WT_E / 2], sd1[WT_E / 2];
    __shared__ float sa2[WT_E / 4], sd2[WT_E / 4];
    __shared__ float sa3[WT_E / 8], sd3[WT_E / 8];
    __shared__ float sr2[WT_E / 4];
    __shared__ float sr1[WT_E / 2];

    const int tid = threadIdx.x;
    const int c = blockIdx.y;
    const int b = blockIdx.z;
    const long rowoff = ((long)b * CC + c) * LL;
    const int l0 = blockIdx.x * WT_T;

    float wc0[5], wc1[5], wc2[5];
#pragma unroll
    for (int j = 0; j < 5; ++j) {
        wc0[j] = w0[c * 5 + j];
        wc1[j] = w1[c * 5 + j];
        wc2[j] = w2[c * 5 + j];
    }

    for (int i = tid; i < WT_E; i += 256) {
        int g = l0 - WT_H + i;
        sx[i] = (g >= 0 && g < LL) ? x[rowoff + g] : 0.f;
    }
    __syncthreads();
    // DWT level 1..3 : lo[m]=SQ*(e+o), hi[m]=SQ*(e-o)   (pad=(0,1), stride 2)
    for (int m = tid; m < WT_E / 2; m += 256) {
        float e = sx[2 * m], o = sx[2 * m + 1];
        sa1[m] = SQf * (e + o); sd1[m] = SQf * (e - o);
    }
    __syncthreads();
    for (int m = tid; m < WT_E / 4; m += 256) {
        float e = sa1[2 * m], o = sa1[2 * m + 1];
        sa2[m] = SQf * (e + o); sd2[m] = SQf * (e - o);
    }
    __syncthreads();
    for (int m = tid; m < WT_E / 8; m += 256) {
        float e = sa2[2 * m], o = sa2[2 * m + 1];
        sa3[m] = SQf * (e + o); sd3[m] = SQf * (e - o);
    }
    __syncthreads();
    // IDWT level 3 (hi = conv5(d3, wc0)):
    //   r[2m]   = 2*SQ*lo[m]
    //   r[2m+1] = SQ*(lo[m]+lo[m+1]) + SQ*(h[m]-h[m+1])
    for (int m = tid; m < WT_E / 8; m += 256) {
        const int M = WT_E / 8;
        float lo0 = sa3[m];
        float lo1 = (m + 1 < M) ? sa3[m + 1] : 0.f;
        float h0 = conv5(sd3, m, M, wc0);
        float h1 = (m + 1 < M) ? conv5(sd3, m + 1, M, wc0) : 0.f;
        sr2[2 * m]     = 2.f * SQf * lo0;
        sr2[2 * m + 1] = SQf * (lo0 + lo1) + SQf * (h0 - h1);
    }
    __syncthreads();
    // IDWT level 2
    for (int m = tid; m < WT_E / 4; m += 256) {
        const int M = WT_E / 4;
        float lo0 = sr2[m];
        float lo1 = (m + 1 < M) ? sr2[m + 1] : 0.f;
        float h0 = conv5(sd2, m, M, wc1);
        float h1 = (m + 1 < M) ? conv5(sd2, m + 1, M, wc1) : 0.f;
        sr1[2 * m]     = 2.f * SQf * lo0;
        sr1[2 * m + 1] = SQf * (lo0 + lo1) + SQf * (h0 - h1);
    }
    __syncthreads();
    // IDWT level 1 + store central 2048 samples
    for (int m = tid; m < WT_E / 2; m += 256) {
        const int M = WT_E / 2;
        float lo0 = sr1[m];
        float lo1 = (m + 1 < M) ? sr1[m + 1] : 0.f;
        float h0 = conv5(sd1, m, M, wc2);
        float h1 = (m + 1 < M) ? conv5(sd1, m + 1, M, wc2) : 0.f;
        float r0 = 2.f * SQf * lo0;
        float r1 = SQf * (lo0 + lo1) + SQf * (h0 - h1);
        int i0 = 2 * m, i1 = 2 * m + 1;
        if (i0 >= WT_H && i0 < WT_H + WT_T) rec[rowoff + l0 + i0 - WT_H] = r0;
        if (i1 >= WT_H && i1 < WT_H + WT_T) rec[rowoff + l0 + i1 - WT_H] = r1;
    }
}

// ---------------------------------------------------------------------------
// Main conv as WMMA GEMM.  WG = 256 threads = 8 waves; tile 128(co) x 128(l).
// Wave (wm in 0..1, wn in 0..3) owns 64x32 = 4x2 16x16 f32 accumulators.
// A fragment (16x32 f16, w[k][co][ci] row-major):
//   lane<16 : K = {ci0..ci0+7, ci0+16..ci0+23}  -> two 16B loads
//   lane>=16: K = {ci0+8..+15, ci0+24..+31}
// B fragment (32x16 f16, xT[b][l][ci]): lane = column l (+tap shift),
//   same K split as A: half-wave h holds K = ci0+8h..+7, ci0+16+8h..+7.
// EDGE=false: l-block is interior, tap shift can never leave [0,L) -> fully
// unconditional loads.  EDGE=true: guarded loads with fragment zeroing.
// Epilogue fuses + bias[co] + scale[co]*rec.
// ---------------------------------------------------------------------------
template <bool EDGE>
__global__ void __launch_bounds__(256)
gemm_conv_kernel(const _Float16* __restrict__ xT,
                 const _Float16* __restrict__ wp,
                 const float* __restrict__ bias,
                 const float* __restrict__ scale,
                 const float* __restrict__ rec,
                 float* __restrict__ out,
                 int lblk0, int lblk_stride)
{
    const int lane = threadIdx.x & 31;
    const int wid  = threadIdx.x >> 5;
    const int wm   = wid & 1;
    const int wn   = wid >> 1;
    const int hl   = lane >> 4;     // half-wave
    const int l16  = lane & 15;
    const int b    = blockIdx.z;
    const int coW  = blockIdx.y * 128 + wm * 64;
    const int lW   = (lblk0 + blockIdx.x * lblk_stride) * 128 + wn * 32;

    const v8f zf = {0.f, 0.f, 0.f, 0.f, 0.f, 0.f, 0.f, 0.f};
    v8f acc[4][2];
#pragma unroll
    for (int mi = 0; mi < 4; ++mi)
#pragma unroll
        for (int ni = 0; ni < 2; ++ni) acc[mi][ni] = zf;

    for (int k = 0; k < 5; ++k) {
        const _Float16* wk = wp + (size_t)k * (CC * CC);
        const int shift = k - 2;
#pragma unroll 2
        for (int ci0 = 0; ci0 < CC; ci0 += 32) {
            V16 a[4];
#pragma unroll
            for (int mi = 0; mi < 4; ++mi) {
                const _Float16* ap =
                    wk + (size_t)(coW + mi * 16 + l16) * CC + ci0 + hl * 8;
                a[mi].h[0] = *(const v8h*)(ap);
                a[mi].h[1] = *(const v8h*)(ap + 16);
            }
            V16 bf[2];
#pragma unroll
            for (int ni = 0; ni < 2; ++ni) {
                const int col = lW + ni * 16 + l16 + shift;
                if (!EDGE || (col >= 0 && col < LL)) {
                    const _Float16* bp =
                        xT + ((size_t)b * LL + col) * CC + ci0 + hl * 8;
                    bf[ni].h[0] = *(const v8h*)(bp);
                    bf[ni].h[1] = *(const v8h*)(bp + 16);
                } else {
#pragma unroll
                    for (int j = 0; j < 8; ++j) {
                        bf[ni].h[0][j] = (_Float16)0.f;
                        bf[ni].h[1][j] = (_Float16)0.f;
                    }
                }
            }
#pragma unroll
            for (int mi = 0; mi < 4; ++mi)
#pragma unroll
                for (int ni = 0; ni < 2; ++ni)
                    acc[mi][ni] = __builtin_amdgcn_wmma_f32_16x16x32_f16(
                        false, a[mi].v, false, bf[ni].v,
                        (short)0, acc[mi][ni], false, false);
        }
    }

    // Epilogue: C layout = VGPR j -> M = j (lanes 0-15) / j+8 (lanes 16-31),
    //           N = lane & 15.
#pragma unroll
    for (int mi = 0; mi < 4; ++mi) {
#pragma unroll
        for (int ni = 0; ni < 2; ++ni) {
            const int co_b = coW + mi * 16 + hl * 8;
            const int l    = lW + ni * 16 + l16;
#pragma unroll
            for (int j = 0; j < 8; ++j) {
                const int co = co_b + j;
                const size_t idx = ((size_t)b * CC + co) * LL + l;
                out[idx] = acc[mi][ni][j] + bias[co] + scale[co] * rec[idx];
            }
        }
    }
}

// ---------------------------------------------------------------------------
extern "C" void kernel_launch(void* const* d_in, const int* in_sizes, int n_in,
                              void* d_out, int out_size, void* d_ws,
                              size_t ws_size, hipStream_t stream)
{
    const float* x      = (const float*)d_in[0];
    const float* conv_w = (const float*)d_in[1];
    const float* conv_b = (const float*)d_in[2];
    const float* wconv0 = (const float*)d_in[3];
    const float* wconv1 = (const float*)d_in[4];
    const float* wconv2 = (const float*)d_in[5];
    const float* scale  = (const float*)d_in[6];
    float* out = (float*)d_out;

    char* ws = (char*)d_ws;
    _Float16* xT = (_Float16*)ws;                           // 64 MiB
    _Float16* wp = (_Float16*)(ws + (size_t)67108864);      // 640 KiB
    float*    rec = (float*)(ws + (size_t)67764224);        // 128 MiB

    pack_w_kernel<<<dim3(1280), 256, 0, stream>>>(conv_w, wp);
    pack_x_kernel<<<dim3(LL / 32, CC / 32, BB), 256, 0, stream>>>(x, xT);
    wavelet_rec_kernel<<<dim3(LL / WT_T, CC, BB), 256, 0, stream>>>(
        x, wconv0, wconv1, wconv2, rec);

    // Interior l-blocks 1..126 : unconditional loads.
    gemm_conv_kernel<false><<<dim3(LL / 128 - 2, CC / 128, BB), 256, 0,
                              stream>>>(xT, wp, conv_b, scale, rec, out,
                                        /*lblk0=*/1, /*stride=*/1);
    // Edge l-blocks {0, 127} : guarded loads.
    gemm_conv_kernel<true><<<dim3(2, CC / 128, BB), 256, 0, stream>>>(
        xT, wp, conv_b, scale, rec, out,
        /*lblk0=*/0, /*stride=*/LL / 128 - 1);
}
